// Kernel_11344467299061915904_53472342835846
// MI455X (gfx1250) — matrix operand order repaired and checked by reading, not compiled, not run on hardware
//
#include <hip/hip_runtime.h>

// ---------------------------------------------------------------------------
// MI455X (gfx1250) implementation.
// All heavy math runs through v_wmma_f32_16x16x32_bf16 (bf16 operands, f32
// accumulate). Problem is compute-bound (~135 GFLOP vs ~350 MB of traffic).
// Round 3: (a) guard-padded im2col buffer removes all per-step predication /
// zero-fill / WMMA-hazard NOPs from the t7 inner loop; (b) 4Mx2S register
// blocking in t7/t9 -> 8 WMMAs per 12 b128 fragment loads.
// ---------------------------------------------------------------------------

typedef __bf16 bf16_t;
typedef __attribute__((ext_vector_type(16))) __bf16 v16bf;
typedef __attribute__((ext_vector_type(8)))  float  v8f;

#define NB 16
#define CC 128
#define HH 64
#define WW 64
#define SS 4096   /* H*W  */
#define K7 896    /* 7*C  */
#define WP 70     /* padded W: 3 guard cols each side */

// ----------------------------- WMMA fragment helpers -----------------------
// CDNA5 ISA 7.12.2 layouts (wave32):
//  A 16x32 bf16 : lane L holds row m=L&15; lanes 0-15: VGPR0..3 K=0..7,
//                 VGPR4..7 K=16..23; lanes 16-31 add +8 to K.
//  B 32x16 bf16 : lane L holds column n=L&15; this code loads lanes 0-15 K=0..15,
//                 lanes 16-31 K=16..31 (contiguous K per lane, A loaded the same way).
//  C/D f32 16x16: element r -> row m = r + (lane>=16 ? 8 : 0), col = lane&15.

__device__ inline v8f f_zero8() {
  v8f v;
#pragma unroll
  for (int i = 0; i < 8; ++i) v[i] = 0.0f;
  return v;
}

// A fragment from row-major [m][k] source (row stride ldk elements).
__device__ inline v16bf load_a_frag(const bf16_t* __restrict__ src, int ldk, int lane) {
  const bf16_t* row = src + (size_t)(lane & 15) * ldk + ((lane & 16) ? 16 : 0);
  v16bf a;
#pragma unroll
  for (int e = 0; e < 16; ++e) {
    int k = e;
    a[e] = row[k];
  }
  return a;
}

// B fragment from B^T layout [n][k] (k contiguous, row stride ldk elements).
__device__ inline v16bf load_bT_frag(const bf16_t* __restrict__ srcT, int ldk, int lane) {
  const bf16_t* row = srcT + (size_t)(lane & 15) * ldk + ((lane & 16) ? 16 : 0);
  v16bf b;
#pragma unroll
  for (int e = 0; e < 16; ++e) b[e] = row[e];
  return b;
}

// B fragment where each lane supplies its own row pointer (k contiguous).
__device__ inline v16bf load_bT_row(const bf16_t* __restrict__ row) {
  v16bf b;
#pragma unroll
  for (int e = 0; e < 16; ++e) b[e] = row[e];
  return b;
}

// ----------------------------- pack kernels --------------------------------

// xb  [n][c][s]        = bf16(x)          (A operand of t8)
// yb  [n][c][s]        = bf16(p2[c,w]*x)  (B^T operand of t8)
// xTp [n][h][w+3][c]   = bf16(x)          (guard-padded B^T operand of t7)
__global__ void pack_x_kernel(const float* __restrict__ x, const float* __restrict__ p2,
                              bf16_t* __restrict__ xb, bf16_t* __restrict__ yb,
                              bf16_t* __restrict__ xTp) {
  size_t idx = (size_t)blockIdx.x * blockDim.x + threadIdx.x;   // < NB*CC*SS
  int w = (int)(idx & 63);
  int h = (int)((idx >> 6) & 63);
  int c = (int)((idx >> 12) & 127);
  int n = (int)(idx >> 19);
  float xv = x[idx];
  xb[idx] = (bf16_t)xv;
  yb[idx] = (bf16_t)(xv * p2[c * WW + w]);
  xTp[(((size_t)n * HH + h) * WP + (w + 3)) * CC + c] = (bf16_t)xv;
}

// Zero the 6 guard columns (w' in {0,1,2,67,68,69}) of xTp.
__global__ void pad_zero_kernel(bf16_t* __restrict__ xTp) {
  int idx = blockIdx.x * blockDim.x + threadIdx.x;              // < NB*HH*6*CC
  if (idx >= NB * HH * 6 * CC) return;
  int c = idx & 127;
  int t = (idx >> 7) % 6;
  int row = idx / (6 * CC);                                     // n*HH + h
  int wp = (t < 3) ? t : (64 + t);                              // 0,1,2,67,68,69
  xTp[((size_t)row * WP + wp) * CC + c] = (bf16_t)0.0f;
}

// w7 (896,128,1,7) f32  ->  w7b [j][896][128] bf16 (c contiguous: A of t7)
__global__ void pack_w7_kernel(const float* __restrict__ w7, bf16_t* __restrict__ w7b) {
  int idx = blockIdx.x * blockDim.x + threadIdx.x;              // < 896*128*7
  if (idx >= K7 * CC * 7) return;
  int j = idx % 7;
  int c = (idx / 7) % CC;
  int m = idx / (7 * CC);
  w7b[((size_t)j * K7 + m) * CC + c] = (bf16_t)w7[idx];
}

// t6 (depthwise 3-tap, dilation 3 in H, after W-roll by 1) written into d_out;
// final GEMM epilogue computes out = t9 - out.
__global__ void t6_kernel(const float* __restrict__ x, const float* __restrict__ p4,
                          const float* __restrict__ w6, float* __restrict__ out) {
  size_t idx = (size_t)blockIdx.x * blockDim.x + threadIdx.x;   // < NB*CC*SS
  int w = (int)(idx & 63);
  int h = (int)((idx >> 6) & 63);
  int c = (int)((idx >> 12) & 127);
  int n = (int)(idx >> 19);
  int w1 = (w + 63) & 63;                                       // roll(+1) source
  float acc = 0.0f;
#pragma unroll
  for (int i = 0; i < 3; ++i) {
    int hh = h + 3 * i - 3;
    if (hh >= 0 && hh < HH) {
      size_t src = (((size_t)n * CC + c) * HH + hh) * WW + w1;
      acc += w6[c * 3 + i] * p4[((size_t)c * HH + hh) * WW + w1] * x[src];
    }
  }
  out[idx] = acc;
}

// ----------------------------- t7 GEMM -------------------------------------
// t7[n,K,s] = sum_{c,j} w7[K,c,j] * x[n,c,h,w+j-3]   (zero-pad in w via xTp)
// One wave per 64x32 block (4 M-tiles x 2 S-tiles): per K-step 4 A + 2 B
// fragments feed 8 WMMAs, fully branchless thanks to guard padding.
// D stored transposed: t7bT[n][s][K]  (K contiguous -> B^T operand of t9).
__global__ void __launch_bounds__(256) t7_gemm_kernel(const bf16_t* __restrict__ w7b,
                                                      const bf16_t* __restrict__ xTp,
                                                      bf16_t* __restrict__ t7bT) {
  int lane = threadIdx.x & 31;
  int wid  = blockIdx.x * 8 + (threadIdx.x >> 5);   // < 16*14*128
  int spair = wid & 127;
  int mquad = (wid >> 7) % 14;
  int n     = wid / (14 * 128);
  int M0 = mquad * 64;
  int S0 = spair * 32;

  v8f acc0a = f_zero8(), acc1a = f_zero8(), acc2a = f_zero8(), acc3a = f_zero8();
  v8f acc0b = f_zero8(), acc1b = f_zero8(), acc2b = f_zero8(), acc3b = f_zero8();

  int s0 = S0 + (lane & 15);
  int s1 = s0 + 16;
  // Padded per-lane row bases: element for tap j, channel c lives at +j*CC+c.
  const bf16_t* bBase0 = xTp + (((size_t)n * HH + (s0 >> 6)) * WP + (s0 & 63)) * CC
                             + ((lane & 16) ? 16 : 0);
  const bf16_t* bBase1 = xTp + (((size_t)n * HH + (s1 >> 6)) * WP + (s1 & 63)) * CC
                             + ((lane & 16) ? 16 : 0);

#pragma unroll
  for (int j = 0; j < 7; ++j) {
    const bf16_t* aBase = w7b + ((size_t)j * K7 + M0) * CC;
#pragma unroll
    for (int c0 = 0; c0 < CC; c0 += 32) {
      v16bf b0 = load_bT_row(bBase0 + (size_t)j * CC + c0);
      v16bf b1 = load_bT_row(bBase1 + (size_t)j * CC + c0);
      v16bf a0 = load_a_frag(aBase + (size_t)0  * CC + c0, CC, lane);
      v16bf a1 = load_a_frag(aBase + (size_t)16 * CC + c0, CC, lane);
      v16bf a2 = load_a_frag(aBase + (size_t)32 * CC + c0, CC, lane);
      v16bf a3 = load_a_frag(aBase + (size_t)48 * CC + c0, CC, lane);
      acc0a = __builtin_amdgcn_wmma_f32_16x16x32_bf16(false, a0, false, b0, (short)0, acc0a, false, false);
      acc1a = __builtin_amdgcn_wmma_f32_16x16x32_bf16(false, a1, false, b0, (short)0, acc1a, false, false);
      acc2a = __builtin_amdgcn_wmma_f32_16x16x32_bf16(false, a2, false, b0, (short)0, acc2a, false, false);
      acc3a = __builtin_amdgcn_wmma_f32_16x16x32_bf16(false, a3, false, b0, (short)0, acc3a, false, false);
      acc0b = __builtin_amdgcn_wmma_f32_16x16x32_bf16(false, a0, false, b1, (short)0, acc0b, false, false);
      acc1b = __builtin_amdgcn_wmma_f32_16x16x32_bf16(false, a1, false, b1, (short)0, acc1b, false, false);
      acc2b = __builtin_amdgcn_wmma_f32_16x16x32_bf16(false, a2, false, b1, (short)0, acc2b, false, false);
      acc3b = __builtin_amdgcn_wmma_f32_16x16x32_bf16(false, a3, false, b1, (short)0, acc3b, false, false);
    }
  }

  const v8f* accA[4] = {&acc0a, &acc1a, &acc2a, &acc3a};
  const v8f* accB[4] = {&acc0b, &acc1b, &acc2b, &acc3b};
  bf16_t* dst0 = t7bT + ((size_t)n * SS + s0) * K7 + M0 + ((lane & 16) ? 8 : 0);
  bf16_t* dst1 = t7bT + ((size_t)n * SS + s1) * K7 + M0 + ((lane & 16) ? 8 : 0);
#pragma unroll
  for (int mt = 0; mt < 4; ++mt) {
#pragma unroll
    for (int r = 0; r < 8; ++r) {
      dst0[mt * 16 + r] = (bf16_t)((*accA[mt])[r]);
      dst1[mt * 16 + r] = (bf16_t)((*accB[mt])[r]);
    }
  }
}

// ----------------------------- t8 GEMM -------------------------------------
// Gk[n,c,c'] = sum_s x[n,c,s] * y[n,c',s + (2k-6)*64]   (y = p2*x, zero-pad h)
// t8[n,c,c'*7+k] = p3[c',k]/64 * Gk. One wave per 32x32 block (2x2 tiles).
__global__ void __launch_bounds__(256) t8_gemm_kernel(const bf16_t* __restrict__ xb,
                                                      const bf16_t* __restrict__ yb,
                                                      const float*  __restrict__ p3,
                                                      bf16_t* __restrict__ t8b) {
  int lane = threadIdx.x & 31;
  int wid  = blockIdx.x * 8 + (threadIdx.x >> 5);                // < 16*7*16
  int nt = wid & 3;                 // c' 32-block
  int ct = (wid >> 2) & 3;          // c  32-block
  int kk = (wid >> 4) % 7;
  int n  = wid / (7 * 16);
  int M0 = ct * 32;
  int N0 = nt * 32;
  int dh = 2 * kk - 6;

  v8f acc00 = f_zero8(), acc01 = f_zero8(), acc10 = f_zero8(), acc11 = f_zero8();
  const bf16_t* aBase = xb + ((size_t)n * CC + M0) * SS;
  const bf16_t* bBase = yb + ((size_t)n * CC + N0) * SS + (ptrdiff_t)dh * WW;

  for (int s0 = 0; s0 < SS; s0 += 32) {     // 32-block stays inside one h-row
    int hh = (s0 >> 6) + dh;
    if (hh < 0 || hh >= HH) continue;       // zero-padded rows: skip (uniform)
    __builtin_prefetch(aBase + s0 + 128, 0, 3);
    __builtin_prefetch(bBase + s0 + 128, 0, 3);
    v16bf a0 = load_a_frag (aBase + s0,                    SS, lane);
    v16bf a1 = load_a_frag (aBase + (size_t)16 * SS + s0,  SS, lane);
    v16bf b0 = load_bT_frag(bBase + s0,                    SS, lane);
    v16bf b1 = load_bT_frag(bBase + (size_t)16 * SS + s0,  SS, lane);
    acc00 = __builtin_amdgcn_wmma_f32_16x16x32_bf16(false, a0, false, b0, (short)0, acc00, false, false);
    acc01 = __builtin_amdgcn_wmma_f32_16x16x32_bf16(false, a0, false, b1, (short)0, acc01, false, false);
    acc10 = __builtin_amdgcn_wmma_f32_16x16x32_bf16(false, a1, false, b0, (short)0, acc10, false, false);
    acc11 = __builtin_amdgcn_wmma_f32_16x16x32_bf16(false, a1, false, b1, (short)0, acc11, false, false);
  }

  const v8f* accs[2][2] = {{&acc00, &acc01}, {&acc10, &acc11}};
#pragma unroll
  for (int i = 0; i < 2; ++i) {
#pragma unroll
    for (int jj = 0; jj < 2; ++jj) {
      int cp = N0 + jj * 16 + (lane & 15);                 // c'
      float scale = p3[cp * 7 + kk] * (1.0f / 64.0f);      // p3 * 1/sqrt(S)
      int mbase = M0 + i * 16 + ((lane & 16) ? 8 : 0);
#pragma unroll
      for (int r = 0; r < 8; ++r) {
        t8b[((size_t)n * CC + (mbase + r)) * K7 + cp * 7 + kk] =
            (bf16_t)((*accs[i][jj])[r] * scale);
      }
    }
  }
}

// ----------------------------- t9 GEMM + epilogue ---------------------------
// t9[n,c,s] = (1/sqrt(896)) * sum_K t8[n,c,K] * t7[n,K,s] ; out = t9 - t6.
// One wave per 64x32 block (4 M-tiles x 2 S-tiles): 8 WMMAs per K-step from
// 4 A + 2 B fragments.
__global__ void __launch_bounds__(256) t9_gemm_kernel(const bf16_t* __restrict__ t8b,
                                                      const bf16_t* __restrict__ t7bT,
                                                      float* __restrict__ out) {
  int lane = threadIdx.x & 31;
  int wid  = blockIdx.x * 8 + (threadIdx.x >> 5);                // < 16*2*128
  int spair = wid & 127;
  int mhalf = (wid >> 7) & 1;
  int n     = wid >> 8;
  int M0 = mhalf * 64;
  int S0 = spair * 32;

  v8f acc0a = f_zero8(), acc1a = f_zero8(), acc2a = f_zero8(), acc3a = f_zero8();
  v8f acc0b = f_zero8(), acc1b = f_zero8(), acc2b = f_zero8(), acc3b = f_zero8();
  const bf16_t* aBase  = t8b  + ((size_t)n * CC + M0) * K7;
  const bf16_t* bBase0 = t7bT + ((size_t)n * SS + S0) * K7;
  const bf16_t* bBase1 = bBase0 + (size_t)16 * K7;

#pragma unroll 2
  for (int k0 = 0; k0 < K7; k0 += 32) {                          // 28 steps
    __builtin_prefetch(bBase0 + k0 + 128, 0, 3);
    __builtin_prefetch(bBase1 + k0 + 128, 0, 3);
    v16bf b0 = load_bT_frag(bBase0 + k0, K7, lane);
    v16bf b1 = load_bT_frag(bBase1 + k0, K7, lane);
    v16bf a0 = load_a_frag (aBase + (size_t)0  * K7 + k0, K7, lane);
    v16bf a1 = load_a_frag (aBase + (size_t)16 * K7 + k0, K7, lane);
    v16bf a2 = load_a_frag (aBase + (size_t)32 * K7 + k0, K7, lane);
    v16bf a3 = load_a_frag (aBase + (size_t)48 * K7 + k0, K7, lane);
    acc0a = __builtin_amdgcn_wmma_f32_16x16x32_bf16(false, a0, false, b0, (short)0, acc0a, false, false);
    acc1a = __builtin_amdgcn_wmma_f32_16x16x32_bf16(false, a1, false, b0, (short)0, acc1a, false, false);
    acc2a = __builtin_amdgcn_wmma_f32_16x16x32_bf16(false, a2, false, b0, (short)0, acc2a, false, false);
    acc3a = __builtin_amdgcn_wmma_f32_16x16x32_bf16(false, a3, false, b0, (short)0, acc3a, false, false);
    acc0b = __builtin_amdgcn_wmma_f32_16x16x32_bf16(false, a0, false, b1, (short)0, acc0b, false, false);
    acc1b = __builtin_amdgcn_wmma_f32_16x16x32_bf16(false, a1, false, b1, (short)0, acc1b, false, false);
    acc2b = __builtin_amdgcn_wmma_f32_16x16x32_bf16(false, a2, false, b1, (short)0, acc2b, false, false);
    acc3b = __builtin_amdgcn_wmma_f32_16x16x32_bf16(false, a3, false, b1, (short)0, acc3b, false, false);
  }

  const float sc = 0.03340766f;                                  // 1/sqrt(896)
  const v8f* accA[4] = {&acc0a, &acc1a, &acc2a, &acc3a};
  const v8f* accB[4] = {&acc0b, &acc1b, &acc2b, &acc3b};
  int sOut0 = S0 + (lane & 15);
#pragma unroll
  for (int mt = 0; mt < 4; ++mt) {
    int mbase = M0 + mt * 16 + ((lane & 16) ? 8 : 0);
#pragma unroll
    for (int r = 0; r < 8; ++r) {
      size_t idx0 = ((size_t)n * CC + (mbase + r)) * SS + sOut0;
      out[idx0]      = (*accA[mt])[r] * sc - out[idx0];          // t9 - t6
      out[idx0 + 16] = (*accB[mt])[r] * sc - out[idx0 + 16];
    }
  }
}

// ----------------------------- launcher ------------------------------------
extern "C" void kernel_launch(void* const* d_in, const int* in_sizes, int n_in,
                              void* d_out, int out_size, void* d_ws, size_t ws_size,
                              hipStream_t stream) {
  const float* x  = (const float*)d_in[0];
  const float* p2 = (const float*)d_in[1];
  const float* p3 = (const float*)d_in[2];
  const float* p4 = (const float*)d_in[3];
  const float* w6 = (const float*)d_in[4];
  const float* w7 = (const float*)d_in[5];
  float* out = (float*)d_out;

  // Workspace layout (bf16 elements):
  //   xb   [16][128][4096]      16.78 MB
  //   yb   [16][128][4096]      16.78 MB
  //   xTp  [16][64][70][128]    18.35 MB  (guard-padded im2col operand)
  //   t7bT [16][4096][896]     117.44 MB
  //   t8b  [16][128][896]        3.67 MB
  //   w7b  [7][896][128]         1.61 MB     total ~175 MB
  char* base = (char*)d_ws;
  bf16_t* xb   = (bf16_t*)base;
  bf16_t* yb   = xb   + (size_t)NB * CC * SS;
  bf16_t* xTp  = yb   + (size_t)NB * CC * SS;
  bf16_t* t7bT = xTp  + (size_t)NB * HH * WP * CC;
  bf16_t* t8b  = t7bT + (size_t)NB * SS * K7;
  bf16_t* w7b  = t8b  + (size_t)NB * CC * K7;

  const int elems = NB * CC * SS;                 // 8,388,608

  pad_zero_kernel<<<(NB * HH * 6 * CC + 255) / 256, 256, 0, stream>>>(xTp);
  pack_x_kernel <<<elems / 256, 256, 0, stream>>>(x, p2, xb, yb, xTp);
  pack_w7_kernel<<<(K7 * CC * 7 + 255) / 256, 256, 0, stream>>>(w7, w7b);
  t6_kernel     <<<elems / 256, 256, 0, stream>>>(x, p4, w6, out);
  t7_gemm_kernel<<<NB * 14 * 128 / 8, 256, 0, stream>>>(w7b, xTp, t7bT);
  t8_gemm_kernel<<<NB * 7 * 16 / 8, 256, 0, stream>>>(xb, yb, p3, t8b);
  t9_gemm_kernel<<<NB * 2 * 128 / 8, 256, 0, stream>>>(t8b, t7bT, out);
}
